// CenMoEDynamicsModel_68358699483593
// MI455X (gfx1250) — matrix operand order repaired and checked by reading, not compiled, not run on hardware
//
#include <hip/hip_runtime.h>
#include <hip/hip_bf16.h>
#include <math.h>

#define D_Z 128
#define D_A 64
#define DD  192
#define HH1 512
#define HH2 512
#define EE  16
#define BB  64
#define NN  2048
#define LN_EPS 1e-5f

typedef __attribute__((ext_vector_type(16))) __bf16 bf16x16;
typedef __attribute__((ext_vector_type(8)))  float  f32x8;
typedef __attribute__((ext_vector_type(2)))  float  f32x2;
typedef __attribute__((ext_vector_type(4)))  unsigned int u32x4;
typedef __attribute__((ext_vector_type(8)))  int  i32x8;
typedef __attribute__((ext_vector_type(4)))  int  i32x4;

#if defined(__gfx1250__) && __has_builtin(__builtin_amdgcn_tensor_load_to_lds) && \
    __has_builtin(__builtin_amdgcn_s_wait_tensorcnt)
#define HAVE_TDM 1
#else
#define HAVE_TDM 0
#endif

__device__ __forceinline__ unsigned short f2bf(float f) {
  unsigned int u = __builtin_bit_cast(unsigned int, f);
  u += 0x7fffu + ((u >> 16) & 1u);           // round-to-nearest-even
  return (unsigned short)(u >> 16);
}
__device__ __forceinline__ __bf16 bfbits(unsigned short u) {
  return __builtin_bit_cast(__bf16, u);
}

__device__ __forceinline__ unsigned ldsAddr(const void* p) {
#if HAVE_TDM
  return (unsigned)(unsigned long long)(__attribute__((address_space(3))) const char*)p;
#else
  return 0;
#endif
}

// 2D TDM tile load: global (row-major, stride_elems between rows) -> LDS packed.
// tile_d0 = elements per row, tile_d1 = rows, ds_code: 0=1B,1=2B,2=4B,3=8B.
__device__ __forceinline__ void tdm_load_2d(unsigned lds_byte_addr, const void* gaddr,
                                            unsigned tile_d0, unsigned tile_d1,
                                            unsigned tensor_d0, unsigned tensor_d1,
                                            unsigned long long stride_elems,
                                            unsigned ds_code) {
#if HAVE_TDM
  unsigned long long ga = (unsigned long long)gaddr;
  u32x4 g0;
  g0[0] = 1u;                                            // count=1, user mode
  g0[1] = lds_byte_addr;                                 // lds_addr [63:32]
  g0[2] = (unsigned)(ga & 0xffffffffu);                  // global_addr lo
  g0[3] = (unsigned)((ga >> 32) & 0x01ffffffu) | (2u << 30);  // addr hi | type=2
  i32x8 g1;
  g1[0] = (int)(ds_code << 16);                          // wg_mask=0, data_size
  g1[1] = (int)((tensor_d0 & 0xffffu) << 16);            // tensor_dim0 lo16 @ [63:48]
  g1[2] = (int)((tensor_d0 >> 16) | ((tensor_d1 & 0xffffu) << 16)); // d0 hi | d1 lo
  g1[3] = (int)((tensor_d1 >> 16) | (tile_d0 << 16));    // d1 hi | tile_dim0 @ [127:112]
  g1[4] = (int)tile_d1;                                  // tile_dim1, tile_dim2=0
  g1[5] = (int)(stride_elems & 0xffffffffu);             // tensor_dim0_stride lo
  g1[6] = (int)((stride_elems >> 32) & 0xffffu);         // stride hi, dim1_stride=0
  g1[7] = 0;
  i32x4 gz4 = {0, 0, 0, 0};
  i32x8 gz8 = {0, 0, 0, 0, 0, 0, 0, 0};
  __builtin_amdgcn_tensor_load_to_lds(g0, g1, gz4, gz4, gz8, 0);  // 6-arg toolchain
#else
  (void)lds_byte_addr; (void)gaddr; (void)tile_d0; (void)tile_d1;
  (void)tensor_d0; (void)tensor_d1; (void)stride_elems; (void)ds_code;
#endif
}

__device__ __forceinline__ void tdm_wait() {
#if HAVE_TDM
  __builtin_amdgcn_s_wait_tensorcnt(0);
#endif
}

// ---------------------------------------------------------------------------
// K1: logits  w[b,n,e] = sum_d x[b,n,d] * phi[d,e]   (x = concat(z,a))
// ---------------------------------------------------------------------------
__global__ void k1_logits(const float* __restrict__ z, const float* __restrict__ a,
                          const float* __restrict__ phi, float* __restrict__ wts) {
  __shared__ float phi_s[DD * EE];
  int tid = threadIdx.x;
  for (int i = tid; i < DD * EE; i += 256) phi_s[i] = phi[i];
  __syncthreads();

  size_t t = (size_t)blockIdx.x * 256 + tid;        // token index in [0, B*N)
  const float4* zr = (const float4*)(z + t * D_Z);
  const float4* ar = (const float4*)(a + t * D_A);

  float w[EE];
#pragma unroll
  for (int e = 0; e < EE; e++) w[e] = 0.f;

  for (int i = 0; i < D_Z / 4; i++) {
    float4 v = zr[i];
    int d4 = i * 4;
#pragma unroll
    for (int e = 0; e < EE; e++)
      w[e] = fmaf(v.x, phi_s[(d4 + 0) * EE + e],
             fmaf(v.y, phi_s[(d4 + 1) * EE + e],
             fmaf(v.z, phi_s[(d4 + 2) * EE + e],
             fmaf(v.w, phi_s[(d4 + 3) * EE + e], w[e]))));
  }
  for (int i = 0; i < D_A / 4; i++) {
    float4 v = ar[i];
    int d4 = D_Z + i * 4;
#pragma unroll
    for (int e = 0; e < EE; e++)
      w[e] = fmaf(v.x, phi_s[(d4 + 0) * EE + e],
             fmaf(v.y, phi_s[(d4 + 1) * EE + e],
             fmaf(v.z, phi_s[(d4 + 2) * EE + e],
             fmaf(v.w, phi_s[(d4 + 3) * EE + e], w[e]))));
  }
#pragma unroll
  for (int e = 0; e < EE; e++) wts[t * EE + e] = w[e];
}

// ---------------------------------------------------------------------------
// K2a: per-(b,e) softmax stats over N tokens: max and 1/sum(exp)
// ---------------------------------------------------------------------------
__global__ void k2a_stats(const float* __restrict__ wts, float* __restrict__ stats) {
  int be = blockIdx.x;                       // b*EE + e
  int b = be / EE, e = be % EE;
  int tid = threadIdx.x;
  __shared__ float red[256];

  float m = -3.4e38f;
  for (int n = tid; n < NN; n += 256)
    m = fmaxf(m, wts[((size_t)b * NN + n) * EE + e]);
  red[tid] = m; __syncthreads();
  for (int s = 128; s > 0; s >>= 1) { if (tid < s) red[tid] = fmaxf(red[tid], red[tid + s]); __syncthreads(); }
  m = red[0]; __syncthreads();

  float sum = 0.f;
  for (int n = tid; n < NN; n += 256)
    sum += expf(wts[((size_t)b * NN + n) * EE + e] - m);
  red[tid] = sum; __syncthreads();
  for (int s = 128; s > 0; s >>= 1) { if (tid < s) red[tid] += red[tid + s]; __syncthreads(); }
  if (tid == 0) { stats[2 * be] = m; stats[2 * be + 1] = 1.f / red[0]; }
}

// ---------------------------------------------------------------------------
// K2b: xin[e][b][d] = sum_n softmax_n(w[b,n,e]) * x[b,n,d]   -> bf16 A-matrix
// ---------------------------------------------------------------------------
__global__ void k2b_xin(const float* __restrict__ z, const float* __restrict__ a,
                        const float* __restrict__ wts, const float* __restrict__ stats,
                        unsigned short* __restrict__ xin_bf) {
  int b = blockIdx.x;
  int tid = threadIdx.x;                     // 256 threads, d = tid (<192 active)
  __shared__ float disp[32 * EE];
  __shared__ float mx[EE], inv[EE];
  if (tid < EE) { mx[tid] = stats[2 * (b * EE + tid)]; inv[tid] = stats[2 * (b * EE + tid) + 1]; }
  __syncthreads();

  float acc[EE];
#pragma unroll
  for (int e = 0; e < EE; e++) acc[e] = 0.f;

  int d = tid;
  for (int n0 = 0; n0 < NN; n0 += 32) {
    for (int i = tid; i < 32 * EE; i += 256) {
      int nn = i / EE, e = i % EE;
      float w = wts[((size_t)b * NN + n0 + nn) * EE + e];
      disp[i] = expf(w - mx[e]) * inv[e];
    }
    __syncthreads();
    if (d < DD) {
      for (int nn = 0; nn < 32; nn++) {
        float xv = (d < D_Z) ? z[((size_t)b * NN + n0 + nn) * D_Z + d]
                             : a[((size_t)b * NN + n0 + nn) * D_A + (d - D_Z)];
#pragma unroll
        for (int e = 0; e < EE; e++) acc[e] = fmaf(disp[nn * EE + e], xv, acc[e]);
      }
    }
    __syncthreads();
  }
  if (d < DD) {
#pragma unroll
    for (int e = 0; e < EE; e++)
      xin_bf[((size_t)e * BB + b) * DD + d] = f2bf(acc[e]);
  }
}

// ---------------------------------------------------------------------------
// K3: per-expert GEMM  Out[e] = A[e](64xK, bf16) * W[e](KxN, f32) + bias
// one block = one (e, n-tile); 4 waves = 4 m-tiles.
// B panel [K x 16] fp32 is TDM-DMA'd into LDS once (32KB max), then the inner
// loop reads B from LDS, A from global: v_wmma_f32_16x16x32_bf16, bf16 conversion on read.
// ---------------------------------------------------------------------------
__global__ void k3_gemm(const unsigned short* __restrict__ Abf,  // [E][64][K] bf16
                        const float* __restrict__ W,             // [E][K][N]  f32
                        const float* __restrict__ bias,          // [E][N]
                        float* __restrict__ Out,                 // [E][64][N]
                        int K, int N) {
  const int ntiles = N >> 4;
  int e  = blockIdx.x / ntiles;
  int nt = blockIdx.x % ntiles;
  int tid  = threadIdx.x;                    // 128 threads
  int lane = tid & 31;
  int wave = tid >> 5;                       // m-tile id (0..3)
  __shared__ float Bs[HH1 * 16];             // worst case 512x16 fp32 = 32KB

  const unsigned short* Ae = Abf + (size_t)e * 64 * K;
  const float*          We = W + (size_t)e * K * N + nt * 16;  // tile start

#if HAVE_TDM
  if (tid < 32) {                            // wave 0 issues the tensor DMA
    tdm_load_2d(ldsAddr(Bs), We, /*tile*/16, (unsigned)K,
                /*tensor*/16, (unsigned)K, (unsigned long long)N, /*4B*/2);
    tdm_wait();
  }
#else
  for (int i = tid; i < K * 16; i += 128)
    Bs[i] = We[(size_t)(i >> 4) * N + (i & 15)];
#endif

  int col  = nt * 16 + (lane & 15);
  int half = lane >> 4;
  int row  = wave * 16 + (lane & 15);
  int koff = half * 8;

  f32x8 acc;
  float bval = bias[e * N + col];
#pragma unroll
  for (int i = 0; i < 8; i++) acc[i] = bval;

  __syncthreads();

  for (int kb = 0; kb < K; kb += 32) {
    bf16x16 afrag, bfrag;
    const unsigned short* arow = Ae + (size_t)row * K + kb;
#pragma unroll
    for (int j = 0; j < 8; j++) {            // A: lanes 0-15 K 0-7/16-23; 16-31 K 8-15/24-31
      afrag[j]     = bfbits(arow[koff + j]);
      afrag[8 + j] = bfbits(arow[16 + koff + j]);
    }
#pragma unroll
    for (int j = 0; j < 16; j++)             // B: lanes 0-15 K 0-7/16-23; 16-31 K 8-15/24-31
      bfrag[j] = bfbits(f2bf(Bs[(kb + half * 8 + (j & 7) + (j >> 3) * 16) * 16 + (lane & 15)]));

    acc = __builtin_amdgcn_wmma_f32_16x16x32_bf16(false, afrag, false, bfrag,
                                                  (short)0, acc, false, false);
  }
  float* oe = Out + (size_t)e * 64 * N;
#pragma unroll
  for (int i = 0; i < 8; i++)                // D: lane col, rows half*8 + i
    oe[(size_t)(wave * 16 + half * 8 + i) * N + col] = acc[i];
}

// ---------------------------------------------------------------------------
// LN + mish rowwise, output bf16 (next layer's A-matrix)
// ---------------------------------------------------------------------------
__global__ void k_lnmish(const float* __restrict__ Hpre,          // [E][64][H]
                         const float* __restrict__ g, const float* __restrict__ be,
                         unsigned short* __restrict__ Hbf, int H) {
  int rowId = blockIdx.x;                    // e*64 + m
  int e = rowId >> 6;
  int tid = threadIdx.x;
  const float* x = Hpre + (size_t)rowId * H;
  __shared__ float red[256];

  float s = 0.f;
  for (int i = tid; i < H; i += 256) s += x[i];
  red[tid] = s; __syncthreads();
  for (int st = 128; st > 0; st >>= 1) { if (tid < st) red[tid] += red[tid + st]; __syncthreads(); }
  float mean = red[0] / H; __syncthreads();

  float v = 0.f;
  for (int i = tid; i < H; i += 256) { float d = x[i] - mean; v += d * d; }
  red[tid] = v; __syncthreads();
  for (int st = 128; st > 0; st >>= 1) { if (tid < st) red[tid] += red[tid + st]; __syncthreads(); }
  float rstd = rsqrtf(red[0] / H + LN_EPS);

  for (int i = tid; i < H; i += 256) {
    float val = (x[i] - mean) * rstd * g[e * H + i] + be[e * H + i];
    float sp  = (val > 20.f) ? val : log1pf(expf(val));
    Hbf[(size_t)rowId * H + i] = f2bf(val * tanhf(sp));
  }
}

// ---------------------------------------------------------------------------
// K4: out[b,n,:] = softmax_e(w[b,n,:]) @ eo[:,b,:]   via v_wmma_f32_16x16x4_f32
// one block = (b, chunk of 128 tokens); 8 waves = 8 m-tiles.
// Expert-output tile (16x128 fp32) is TDM-DMA'd while waves 0-3 compute the
// per-token softmax into LDS (DMA/compute overlap), then s_wait_tensorcnt.
// ---------------------------------------------------------------------------
__global__ void k4_combine(const float* __restrict__ wts,   // [B][N][16]
                           const float* __restrict__ eo,    // [E][64][128] = [k][b][z]
                           float* __restrict__ out) {       // [B][N][128]
  int b  = blockIdx.x >> 4;
  int n0 = (blockIdx.x & 15) * 128;
  int tid = threadIdx.x;                     // 256 threads
  int lane = tid & 31, wave = tid >> 5;
  __shared__ float comb[128 * 16];
  __shared__ float eos[16 * 128];

#if HAVE_TDM
  if (tid < 32)                              // wave 0: DMA 16 rows of 128 floats
    tdm_load_2d(ldsAddr(eos), eo + (size_t)b * D_Z, /*tile*/D_Z, EE,
                /*tensor*/D_Z, EE, (unsigned long long)BB * D_Z, /*4B*/2);
#else
  for (int i = tid; i < 16 * 128; i += 256) {
    int k = i >> 7, zi = i & 127;
    eos[i] = eo[(size_t)k * BB * D_Z + (size_t)b * D_Z + zi];
  }
#endif

  if (tid < 128) {                           // per-token 16-way softmax (overlaps DMA)
    float w[16]; float m = -3.4e38f;
#pragma unroll
    for (int k = 0; k < 16; k++) { w[k] = wts[((size_t)b * NN + n0 + tid) * EE + k]; m = fmaxf(m, w[k]); }
    float s = 0.f;
#pragma unroll
    for (int k = 0; k < 16; k++) { w[k] = expf(w[k] - m); s += w[k]; }
    float inv = 1.f / s;
#pragma unroll
    for (int k = 0; k < 16; k++) comb[tid * 16 + k] = w[k] * inv;
  }
#if HAVE_TDM
  if (tid < 32) tdm_wait();
#endif
  __syncthreads();

  int mrow = wave * 16 + (lane & 15);        // token within chunk (A row)
  int half = lane >> 4;
  for (int nt = 0; nt < 8; nt++) {
    f32x8 acc;
#pragma unroll
    for (int i = 0; i < 8; i++) acc[i] = 0.f;
    int col = nt * 16 + (lane & 15);
#pragma unroll
    for (int kb = 0; kb < 16; kb += 4) {
      f32x2 af, bf;
      af[0] = comb[mrow * 16 + kb + half * 2 + 0];       // A 16x4 f32 layout
      af[1] = comb[mrow * 16 + kb + half * 2 + 1];
      bf[0] = eos[(kb + half * 2 + 0) * 128 + col];      // B 4x16 f32 layout
      bf[1] = eos[(kb + half * 2 + 1) * 128 + col];
      acc = __builtin_amdgcn_wmma_f32_16x16x4_f32(false, af, false, bf,
                                                  (short)0, acc, false, false);
    }
#pragma unroll
    for (int i = 0; i < 8; i++)
      out[((size_t)b * NN + n0 + wave * 16 + half * 8 + i) * D_Z + col] = acc[i];
  }
}

// ---------------------------------------------------------------------------
extern "C" void kernel_launch(void* const* d_in, const int* in_sizes, int n_in,
                              void* d_out, int out_size, void* d_ws, size_t ws_size,
                              hipStream_t stream) {
  const float* z   = (const float*)d_in[0];
  const float* a   = (const float*)d_in[1];
  const float* phi = (const float*)d_in[2];
  const float* W1  = (const float*)d_in[3];
  const float* b1  = (const float*)d_in[4];
  const float* g1  = (const float*)d_in[5];
  const float* be1 = (const float*)d_in[6];
  const float* W2  = (const float*)d_in[7];
  const float* b2  = (const float*)d_in[8];
  const float* g2  = (const float*)d_in[9];
  const float* be2 = (const float*)d_in[10];
  const float* W3  = (const float*)d_in[11];
  const float* b3  = (const float*)d_in[12];
  float* out = (float*)d_out;
  char* ws = (char*)d_ws;

  size_t off = 0;
  float*          wts    = (float*)(ws + off);          off += (size_t)BB * NN * EE * 4;   // 8 MB
  float*          stats  = (float*)(ws + off);          off += (size_t)BB * EE * 2 * 4;    // 8 KB
  unsigned short* xin_bf = (unsigned short*)(ws + off); off += (size_t)EE * BB * DD * 2;   // 384 KB
  float*          h1pre  = (float*)(ws + off);          off += (size_t)EE * BB * HH1 * 4;  // 2 MB
  unsigned short* h1bf   = (unsigned short*)(ws + off); off += (size_t)EE * BB * HH1 * 2;  // 1 MB
  float*          h2pre  = (float*)(ws + off);          off += (size_t)EE * BB * HH2 * 4;  // 2 MB
  unsigned short* h2bf   = (unsigned short*)(ws + off); off += (size_t)EE * BB * HH2 * 2;  // 1 MB
  float*          eo     = (float*)(ws + off);          off += (size_t)EE * BB * D_Z * 4;  // 512 KB

  k1_logits <<<(BB * NN) / 256, 256, 0, stream>>>(z, a, phi, wts);
  k2a_stats <<<BB * EE,          256, 0, stream>>>(wts, stats);
  k2b_xin   <<<BB,               256, 0, stream>>>(z, a, wts, stats, xin_bf);

  k3_gemm   <<<EE * (HH1 / 16),  128, 0, stream>>>(xin_bf, W1, b1, h1pre, DD,  HH1);
  k_lnmish  <<<EE * BB,          256, 0, stream>>>(h1pre, g1, be1, h1bf, HH1);
  k3_gemm   <<<EE * (HH2 / 16),  128, 0, stream>>>(h1bf,  W2, b2, h2pre, HH1, HH2);
  k_lnmish  <<<EE * BB,          256, 0, stream>>>(h2pre, g2, be2, h2bf, HH2);
  k3_gemm   <<<EE * (D_Z / 16),  128, 0, stream>>>(h2bf,  W3, b3, eo,   HH2, D_Z);

  k4_combine<<<BB * 16,          256, 0, stream>>>(wts, eo, out);
}
